// VigenereKeyPredictor_17308718203264
// MI455X (gfx1250) — matrix operand order repaired and checked by reading, not compiled, not run on hardware
//
#include <hip/hip_runtime.h>
#include <math.h>

typedef __attribute__((ext_vector_type(16))) _Float16 v16h;
typedef __attribute__((ext_vector_type(8)))  float    v8f;

#define WMMA_F32_F16(A, Bf, C) \
  __builtin_amdgcn_wmma_f32_16x16x32_f16(false, (A), false, (Bf), (short)0, (C), false, false)

constexpr int Bb = 512, Ss = 256, Tt = 10, Vv = 28, Ee = 32, Hh = 128, G4 = 512;

__device__ __forceinline__ float sigm(float x) { return 1.f / (1.f + __expf(-x)); }

// A fragment (16 x 32 f16) from LDS row-major [16][ldk] _Float16 array.
// Layout per ISA 7.12.2: lanes 0-15 / 16-31 both hold rows M=0..15;
// half h -> K = (h&7) + 8*laneGrp + 16*(h>>3).
__device__ __forceinline__ v16h load_a_lds(const _Float16* base, int ldk, int k0, int lane) {
  int m = lane & 15, g = lane >> 4;
  v16h a;
#pragma unroll
  for (int h = 0; h < 16; ++h)
    a[h] = base[m * ldk + k0 + (h & 7) + 8 * g + 16 * (h >> 3)];
  return a;
}

// B fragment (32 x 16) from global row-major f32 weight W[n][k] (row stride ldk):
// computes as x @ W^T. B element (k,n): lane = n + 16*((k/8)&1), half = (k&7) + 8*(k/16).
__device__ __forceinline__ v16h load_b_w(const float* W, int ldk, int n0, int k0, int lane) {
  int n = lane & 15, g = lane >> 4;
  const float* r = W + (size_t)(n0 + n) * ldk + k0;
  v16h b;
#pragma unroll
  for (int h = 0; h < 16; ++h) b[h] = (_Float16)r[(h & 7) + 8 * g + 16 * (h >> 3)];
  return b;
}

// C/D fragment scatter to LDS f32 [16][ldn]: vgpr r -> row r + 8*(lane>>4), col = n0 + lane&15.
__device__ __forceinline__ void store_c_lds(float* dst, int ldn, int n0, v8f c, int lane) {
  int n = n0 + (lane & 15), g = lane >> 4;
#pragma unroll
  for (int r = 0; r < 8; ++r) dst[(r + 8 * g) * ldn + n] = c[r];
}

// ---------------- Encoder: one direction, 16 batch rows per workgroup ----------------
__global__ __launch_bounds__(256) void lstm_encoder_kernel(
    const int* __restrict__ src, const float* __restrict__ embed,
    const float* __restrict__ Wih, const float* __restrict__ Whh,
    const float* __restrict__ bias, float* __restrict__ enc_out,
    float* __restrict__ hfin, float* __restrict__ cfin, int reverse) {
  __shared__ _Float16 s_emb[Vv * Ee];
  __shared__ _Float16 s_h[16 * Hh];
  __shared__ float    s_g[16 * G4];

  const int tid = threadIdx.x, lane = tid & 31, wave = tid >> 5;
  const int b0 = blockIdx.x * 16;

  for (int i = tid; i < Vv * Ee; i += 256) s_emb[i] = (_Float16)embed[i];
  for (int i = tid; i < 16 * Hh; i += 256) s_h[i] = (_Float16)0.f;
  float creg[8];
#pragma unroll
  for (int k = 0; k < 8; ++k) creg[k] = 0.f;

  // Weight fragments pinned in VGPRs for all 256 steps (wave owns gate cols 64w..64w+63)
  v16h Bw[4][4], Bx[4];
  float bia[4];
#pragma unroll
  for (int i = 0; i < 4; ++i) {
    int n0 = 64 * wave + 16 * i;
#pragma unroll
    for (int kt = 0; kt < 4; ++kt) Bw[i][kt] = load_b_w(Whh, Hh, n0, 32 * kt, lane);
    Bx[i] = load_b_w(Wih, Ee, n0, 0, lane);
    bia[i] = bias[n0 + (lane & 15)];
  }
  __syncthreads();

  for (int t = 0; t < Ss; ++t) {
    const int pos = reverse ? (Ss - 1 - t) : t;
    const int m = lane & 15, g = lane >> 4;
    const int tok = src[(size_t)(b0 + m) * Ss + pos];
    v16h ae;
#pragma unroll
    for (int h = 0; h < 16; ++h)
      ae[h] = s_emb[tok * Ee + (h & 7) + 8 * g + 16 * (h >> 3)];
    v16h ah[4];
#pragma unroll
    for (int kt = 0; kt < 4; ++kt) ah[kt] = load_a_lds(s_h, Hh, 32 * kt, lane);

#pragma unroll
    for (int i = 0; i < 4; ++i) {
      v8f acc;
#pragma unroll
      for (int r = 0; r < 8; ++r) acc[r] = bia[i];
      acc = WMMA_F32_F16(ae, Bx[i], acc);          // x_t @ Wih^T
#pragma unroll
      for (int kt = 0; kt < 4; ++kt) acc = WMMA_F32_F16(ah[kt], Bw[i][kt], acc);  // h @ Whh^T
      store_c_lds(s_g, G4, 64 * wave + 16 * i, acc, lane);
    }
    __syncthreads();

#pragma unroll
    for (int k = 0; k < 8; ++k) {
      int p = tid + 256 * k, mm = p >> 7, j = p & 127;
      float iv = s_g[mm * G4 + j],        fv = s_g[mm * G4 + Hh + j];
      float gv = s_g[mm * G4 + 2 * Hh + j], ov = s_g[mm * G4 + 3 * Hh + j];
      float cn = sigm(fv) * creg[k] + sigm(iv) * tanhf(gv);
      creg[k] = cn;
      float hn = sigm(ov) * tanhf(cn);
      s_h[mm * Hh + j] = (_Float16)hn;
      enc_out[((size_t)(b0 + mm) * Ss + pos) * (2 * Hh) + (reverse ? Hh : 0) + j] = hn;
      if (t == Ss - 1) {
        hfin[(size_t)(b0 + mm) * Hh + j] = hn;
        cfin[(size_t)(b0 + mm) * Hh + j] = cn;
      }
    }
    __syncthreads();
  }
}

// ---------------- enc_proj = enc_out[B*S,256] @ proj_W^T[256,128] ----------------
__global__ __launch_bounds__(256) void enc_proj_kernel(
    const float* __restrict__ enc_out, const float* __restrict__ proj_W,
    const float* __restrict__ proj_b, float* __restrict__ enc_proj) {
  const int tid = threadIdx.x, lane = tid & 31, wave = tid >> 5;
  const long mtile = (long)blockIdx.x * 8 + wave;   // 8192 tiles of 16 rows
  const long row0 = mtile * 16;
  const int m = lane & 15, g = lane >> 4;
  v16h a[8];
#pragma unroll
  for (int kt = 0; kt < 8; ++kt) {
    const float* rp = enc_out + (row0 + m) * (size_t)(2 * Hh) + 32 * kt;
#pragma unroll
    for (int h = 0; h < 16; ++h) a[kt][h] = (_Float16)rp[(h & 7) + 8 * g + 16 * (h >> 3)];
  }
#pragma unroll
  for (int nt = 0; nt < 8; ++nt) {
    float pb = proj_b[16 * nt + m];
    v8f acc;
#pragma unroll
    for (int r = 0; r < 8; ++r) acc[r] = pb;
#pragma unroll
    for (int kt = 0; kt < 8; ++kt) {
      v16h b = load_b_w(proj_W, 2 * Hh, 16 * nt, 32 * kt, lane);
      acc = WMMA_F32_F16(a[kt], b, acc);
    }
#pragma unroll
    for (int r = 0; r < 8; ++r)
      enc_proj[(row0 + r + 8 * g) * (size_t)Hh + 16 * nt + m] = acc[r];
  }
}

// ---------------- Decoder: LSTM cell + attention + output head, 16 rows / WG ----------------
__global__ __launch_bounds__(256) void decoder_kernel(
    const int* __restrict__ target, const float* __restrict__ embed,
    const float* __restrict__ Wih, const float* __restrict__ Whh, const float* __restrict__ bias,
    const float* __restrict__ attn_W, const float* __restrict__ attn_b,
    const float* __restrict__ out_W, const float* __restrict__ out_b,
    const float* __restrict__ enc_proj,
    const float* __restrict__ hf, const float* __restrict__ cf,
    const float* __restrict__ hb, const float* __restrict__ cb,
    float* __restrict__ out) {
  __shared__ _Float16 s_emb[Vv * Ee];
  __shared__ _Float16 s_hc[16 * 256];   // [16][256]: cols 0..127 = h, 128..255 = ctx (f16)
  __shared__ float    s_g[16 * G4];     // gate pre-acts; reused for comb[16][128]
  __shared__ float    s_sc[16 * Ss];    // attention scores / weights
  __shared__ float    s_red[16 * 16], s_rmax[16], s_rsum[16];

  const int tid = threadIdx.x, lane = tid & 31, wave = tid >> 5;
  const int b0 = blockIdx.x * 16;

  for (int i = tid; i < Vv * Ee; i += 256) s_emb[i] = (_Float16)embed[i];

  float creg[8];
#pragma unroll
  for (int k = 0; k < 8; ++k) {   // h0 = hf+hb, c0 = cf+cb
    int p = tid + 256 * k, mm = p >> 7, j = p & 127;
    size_t off = (size_t)(b0 + mm) * Hh + j;
    creg[k] = cf[off] + cb[off];
    s_hc[mm * 256 + j] = (_Float16)(hf[off] + hb[off]);
  }

  v16h Bw[4][4], Bx[4];
  float bia[4];
#pragma unroll
  for (int i = 0; i < 4; ++i) {
    int n0 = 64 * wave + 16 * i;
#pragma unroll
    for (int kt = 0; kt < 4; ++kt) Bw[i][kt] = load_b_w(Whh, Hh, n0, 32 * kt, lane);
    Bx[i] = load_b_w(Wih, Ee, n0, 0, lane);
    bia[i] = bias[n0 + (lane & 15)];
  }
  const float abn = attn_b[16 * wave + (lane & 15)];
  __syncthreads();

  for (int t = 0; t < Tt; ++t) {
    // ---- LSTM cell ----
    const int m = lane & 15, g = lane >> 4;
    const int tok = (t == 0) ? 0 : target[(size_t)(b0 + m) * Tt + (t - 1)];
    v16h ae;
#pragma unroll
    for (int h = 0; h < 16; ++h) ae[h] = s_emb[tok * Ee + (h & 7) + 8 * g + 16 * (h >> 3)];
    v16h ah[4];
#pragma unroll
    for (int kt = 0; kt < 4; ++kt) ah[kt] = load_a_lds(s_hc, 256, 32 * kt, lane);
#pragma unroll
    for (int i = 0; i < 4; ++i) {
      v8f acc;
#pragma unroll
      for (int r = 0; r < 8; ++r) acc[r] = bia[i];
      acc = WMMA_F32_F16(ae, Bx[i], acc);
#pragma unroll
      for (int kt = 0; kt < 4; ++kt) acc = WMMA_F32_F16(ah[kt], Bw[i][kt], acc);
      store_c_lds(s_g, G4, 64 * wave + 16 * i, acc, lane);
    }
    __syncthreads();
#pragma unroll
    for (int k = 0; k < 8; ++k) {
      int p = tid + 256 * k, mm = p >> 7, j = p & 127;
      float iv = s_g[mm * G4 + j],        fv = s_g[mm * G4 + Hh + j];
      float gv = s_g[mm * G4 + 2 * Hh + j], ov = s_g[mm * G4 + 3 * Hh + j];
      float cn = sigm(fv) * creg[k] + sigm(iv) * tanhf(gv);
      creg[k] = cn;
      s_hc[mm * 256 + j] = (_Float16)(sigm(ov) * tanhf(cn));
    }
    __syncthreads();

    // ---- scores[mm][s] = enc_proj[b][s][:] . h[mm][:]  (thread = s, L2-resident) ----
    {
      const int s = tid;
      for (int mm = 0; mm < 16; ++mm) {
        const float* ep = enc_proj + ((size_t)(b0 + mm) * Ss + s) * Hh;
        __builtin_prefetch(ep, 0, 0);
        float acc = 0.f;
#pragma unroll 4
        for (int k = 0; k < Hh; ++k) acc += ep[k] * (float)s_hc[mm * 256 + k];
        s_sc[mm * Ss + s] = acc;
      }
    }
    __syncthreads();
    // ---- softmax over s (16 threads per row) ----
    {
      int mm = tid >> 4, li = tid & 15;
      float mx = -1e30f;
      for (int s = li; s < Ss; s += 16) mx = fmaxf(mx, s_sc[mm * Ss + s]);
      s_red[mm * 16 + li] = mx;
      __syncthreads();
      if (li == 0) {
        float m2 = -1e30f;
        for (int q = 0; q < 16; ++q) m2 = fmaxf(m2, s_red[mm * 16 + q]);
        s_rmax[mm] = m2;
      }
      __syncthreads();
      float sum = 0.f;
      for (int s = li; s < Ss; s += 16) {
        float e = __expf(s_sc[mm * Ss + s] - s_rmax[mm]);
        s_sc[mm * Ss + s] = e;
        sum += e;
      }
      s_red[mm * 16 + li] = sum;
      __syncthreads();
      if (li == 0) {
        float t2 = 0.f;
        for (int q = 0; q < 16; ++q) t2 += s_red[mm * 16 + q];
        s_rsum[mm] = 1.f / t2;
      }
      __syncthreads();
      for (int s = li; s < Ss; s += 16) s_sc[mm * Ss + s] *= s_rsum[mm];
    }
    __syncthreads();
    // ---- ctx[mm][j] = sum_s aw[s] * enc_proj[b][s][j] ----
#pragma unroll
    for (int k = 0; k < 8; ++k) {
      int p = tid + 256 * k, mm = p >> 7, j = p & 127;
      const float* ep = enc_proj + (size_t)(b0 + mm) * Ss * Hh + j;
      float acc = 0.f;
      for (int s = 0; s < Ss; ++s) acc += s_sc[mm * Ss + s] * ep[(size_t)s * Hh];
      s_hc[mm * 256 + 128 + j] = (_Float16)acc;
    }
    __syncthreads();
    // ---- comb = tanh([h,ctx] @ attn_W^T + b): wave w owns N-tile w ----
    {
      v8f acc;
#pragma unroll
      for (int r = 0; r < 8; ++r) acc[r] = abn;
#pragma unroll
      for (int kt = 0; kt < 8; ++kt) {
        v16h bfrag = load_b_w(attn_W, 2 * Hh, 16 * wave, 32 * kt, lane);
        v16h a = load_a_lds(s_hc, 256, 32 * kt, lane);
        acc = WMMA_F32_F16(a, bfrag, acc);
      }
      int n = 16 * wave + (lane & 15);
#pragma unroll
      for (int r = 0; r < 8; ++r) s_g[(r + 8 * g) * Hh + n] = tanhf(acc[r]);
    }
    __syncthreads();
    // ---- logits = comb @ out_W^T + out_b (V=28, tiny -> VALU) ----
    for (int p = tid; p < 16 * Vv; p += 256) {
      int mm = p / Vv, v = p - mm * Vv;
      float acc = out_b[v];
      const float* w = out_W + (size_t)v * Hh;
#pragma unroll 4
      for (int k = 0; k < Hh; ++k) acc += s_g[mm * Hh + k] * w[k];
      out[((size_t)(b0 + mm) * Tt + t) * Vv + v] = acc;
    }
    __syncthreads();
  }
}

extern "C" void kernel_launch(void* const* d_in, const int* in_sizes, int n_in,
                              void* d_out, int out_size, void* d_ws, size_t ws_size,
                              hipStream_t stream) {
  (void)in_sizes; (void)n_in; (void)out_size; (void)ws_size;
  const int*   src    = (const int*)  d_in[0];
  const int*   target = (const int*)  d_in[1];
  const float* embed  = (const float*)d_in[2];
  const float* eWih_f = (const float*)d_in[3];
  const float* eWhh_f = (const float*)d_in[4];
  const float* eb_f   = (const float*)d_in[5];
  const float* eWih_b = (const float*)d_in[6];
  const float* eWhh_b = (const float*)d_in[7];
  const float* eb_b   = (const float*)d_in[8];
  const float* dWih   = (const float*)d_in[9];
  const float* dWhh   = (const float*)d_in[10];
  const float* db     = (const float*)d_in[11];
  const float* proj_W = (const float*)d_in[12];
  const float* proj_b = (const float*)d_in[13];
  const float* attn_W = (const float*)d_in[14];
  const float* attn_b = (const float*)d_in[15];
  const float* out_W  = (const float*)d_in[16];
  const float* out_b  = (const float*)d_in[17];
  float* out = (float*)d_out;

  float* ws = (float*)d_ws;
  float* enc_out  = ws;                               // B*S*2H f32  (128 MB)
  float* enc_proj = enc_out + (size_t)Bb * Ss * 2 * Hh; // B*S*H f32 (64 MB)
  float* hf = enc_proj + (size_t)Bb * Ss * Hh;
  float* cf = hf + (size_t)Bb * Hh;
  float* hb = cf + (size_t)Bb * Hh;
  float* cb = hb + (size_t)Bb * Hh;

  lstm_encoder_kernel<<<Bb / 16, 256, 0, stream>>>(src, embed, eWih_f, eWhh_f, eb_f,
                                                   enc_out, hf, cf, 0);
  lstm_encoder_kernel<<<Bb / 16, 256, 0, stream>>>(src, embed, eWih_b, eWhh_b, eb_b,
                                                   enc_out, hb, cb, 1);
  enc_proj_kernel<<<(Bb * Ss / 16) / 8, 256, 0, stream>>>(enc_out, proj_W, proj_b, enc_proj);
  decoder_kernel<<<Bb / 16, 256, 0, stream>>>(target, embed, dWih, dWhh, db, attn_W, attn_b,
                                              out_W, out_b, enc_proj, hf, cf, hb, cb, out);
}
